// AdvancedGCN_61272003444817
// MI455X (gfx1250) — compile-verified, matrix operand order checked by reading, not run on hardware
//
#include <hip/hip_runtime.h>
#include <cstdint>

#define NN 50000
#define EE 800000
#define TILES_M 3125   // 50000 / 16 exactly

typedef __attribute__((ext_vector_type(16))) _Float16 v16h;
typedef __attribute__((ext_vector_type(8)))  _Float16 v8h;
typedef __attribute__((ext_vector_type(8)))  float    v8f;

// ---------------- degree / dinv ----------------
__global__ void deg_kernel(const long long* __restrict__ ei, float* __restrict__ deg) {
    int e = blockIdx.x * blockDim.x + threadIdx.x;
    if (e < EE) {
        int d = (int)ei[EE + e];            // dst row of edge_index
        unsafeAtomicAdd(&deg[d], 1.0f);
    }
}

__global__ void dinv_kernel(float* __restrict__ deg) {
    int i = blockIdx.x * blockDim.x + threadIdx.x;
    if (i < NN) deg[i] = rsqrtf(deg[i] + 1.0f);   // +1 self loop
}

// ---------------- weight transpose + f16 convert ----------------
// Wt[n*K + k] = (n < Nout) ? W[k*Nout + n] : 0   — padded to NPAD rows so the
// GEMM K-loop needs no column guard.  K is a power of two (kshift).
__global__ void wt_kernel(const float* __restrict__ W, _Float16* __restrict__ Wt,
                          int K, int kshift, int Nout, int total)
{
    int idx = blockIdx.x * blockDim.x + threadIdx.x;
    if (idx < total) {
        int n = idx >> kshift;
        int k = idx & (K - 1);
        Wt[idx] = (n < Nout) ? (_Float16)W[(size_t)k * Nout + n] : (_Float16)0.0f;
    }
}

// ---------------- WMMA GEMM: C[M,NOUT] = A[M,K] @ W[K,NOUT] (+bias)(+relu) ----
// One wave per 16x16 tile; compile-time shapes -> fully unrolled K loop.
// A: f32 row-major [M,K]; Wt: f16 transposed+padded [NPAD,K].
template <int K, int NOUT, int NPAD, bool BIAS, bool RELU>
__global__ void __launch_bounds__(128)
wmma_gemm(const float* __restrict__ A, const _Float16* __restrict__ Wt,
          const float* __restrict__ bias, float* __restrict__ C)
{
    constexpr int TILES_N = NPAD / 16;
    const int wave  = blockIdx.x * (blockDim.x >> 5) + (threadIdx.x >> 5);
    if (wave >= TILES_M * TILES_N) return;        // wave-uniform

    const int tn   = wave % TILES_N;
    const int tm   = wave / TILES_N;
    const int lane = threadIdx.x & 31;
    const int half = lane >> 4;                   // 0 or 1
    const int l15  = lane & 15;
    const int kb   = half * 8;                    // K sub-offset per ISA A/B layout
    const int mrow = tm * 16 + l15;
    const int ncol = tn * 16 + l15;

    const float*    ar = A  + (size_t)mrow * K + kb;
    const _Float16* br = Wt + (size_t)ncol * K + kb;

    v8f acc = {};
    #pragma unroll
    for (int kc = 0; kc < K; kc += 32) {
        // A fragment: lane holds A[mrow][kc+kb..+7] and A[mrow][kc+16+kb..+7]
        float4 a0 = *(const float4*)(ar + kc);
        float4 a1 = *(const float4*)(ar + kc + 4);
        float4 a2 = *(const float4*)(ar + kc + 16);
        float4 a3 = *(const float4*)(ar + kc + 20);
        v16h av;
        av[0]=(_Float16)a0.x; av[1]=(_Float16)a0.y; av[2]=(_Float16)a0.z; av[3]=(_Float16)a0.w;
        av[4]=(_Float16)a1.x; av[5]=(_Float16)a1.y; av[6]=(_Float16)a1.z; av[7]=(_Float16)a1.w;
        av[8]=(_Float16)a2.x; av[9]=(_Float16)a2.y; av[10]=(_Float16)a2.z; av[11]=(_Float16)a2.w;
        av[12]=(_Float16)a3.x; av[13]=(_Float16)a3.y; av[14]=(_Float16)a3.z; av[15]=(_Float16)a3.w;

        // B fragment: lane holds Wt[ncol][kc+kb..+7] and Wt[ncol][kc+16+kb..+7]
        v8h b0 = *(const v8h*)(br + kc);
        v8h b1 = *(const v8h*)(br + kc + 16);
        v16h bv;
        #pragma unroll
        for (int i = 0; i < 8; ++i) { bv[i] = b0[i]; bv[8 + i] = b1[i]; }

        acc = __builtin_amdgcn_wmma_f32_16x16x32_f16(false, av, false, bv,
                                                     (short)0, acc, false, false);
    }

    float bb = 0.0f;
    if constexpr (BIAS) {
        if constexpr (NOUT == NPAD) bb = bias[ncol];
        else                        bb = (ncol < NOUT) ? bias[ncol] : 0.0f;
    }
    if constexpr (NOUT == NPAD) {
        #pragma unroll
        for (int r = 0; r < 8; ++r) {             // row = tm*16 + half*8 + r
            float v = acc[r] + bb;
            if constexpr (RELU) v = fmaxf(v, 0.0f);
            C[(size_t)(tm * 16 + kb + r) * NOUT + ncol] = v;
        }
    } else {
        if (ncol < NOUT) {
            #pragma unroll
            for (int r = 0; r < 8; ++r) {
                float v = acc[r] + bb;
                if constexpr (RELU) v = fmaxf(v, 0.0f);
                C[(size_t)(tm * 16 + kb + r) * NOUT + ncol] = v;
            }
        }
    }
}

// ---------------- GCN aggregation ----------------
// agg[i,f] = h[i,f]*dinv[i]^2 + bias[f]   (self-loop term + conv bias; zero-inits agg)
__global__ void selfterm_kernel(const float* __restrict__ h, const float* __restrict__ dinv,
                                const float* __restrict__ bias, float* __restrict__ agg,
                                int F, int shift)
{
    int idx = blockIdx.x * blockDim.x + threadIdx.x;
    if (idx < NN * F) {
        int i = idx >> shift;
        int f = idx & (F - 1);
        float di = dinv[i];
        agg[idx] = h[idx] * di * di + bias[f];
    }
}

// agg[dst,f] += h[src,f] * dinv[src]*dinv[dst]  — one thread per (edge, feature)
__global__ void scatter_kernel(const float* __restrict__ h, const float* __restrict__ dinv,
                               const long long* __restrict__ ei, float* __restrict__ agg,
                               int F, int shift)
{
    long long idx = (long long)blockIdx.x * blockDim.x + threadIdx.x;
    long long tot = (long long)EE << shift;
    if (idx < tot) {
        int e = (int)(idx >> shift);
        int f = (int)(idx & (F - 1));
        int s = (int)ei[e];
        int d = (int)ei[EE + e];
        float c = dinv[s] * dinv[d];
        unsafeAtomicAdd(&agg[(size_t)d * F + f], h[(size_t)s * F + f] * c);
    }
}

// ---------------- BatchNorm ----------------
__global__ void bn_stats_kernel(const float* __restrict__ x, float* __restrict__ stats, int F) {
    int f    = threadIdx.x;                       // blockDim.x == F
    int row0 = blockIdx.x * 512;
    int row1 = row0 + 512; if (row1 > NN) row1 = NN;
    float s = 0.0f, q = 0.0f;
    for (int r = row0; r < row1; ++r) {
        float v = x[(size_t)r * F + f];
        s += v; q += v * v;
    }
    unsafeAtomicAdd(&stats[f], s);
    unsafeAtomicAdd(&stats[F + f], q);
}

__global__ void bn_norm_relu_kernel(float* __restrict__ x, const float* __restrict__ stats,
                                    const float* __restrict__ g, const float* __restrict__ be,
                                    int F, int shift)
{
    int idx = blockIdx.x * blockDim.x + threadIdx.x;
    if (idx < NN * F) {
        int f = idx & (F - 1);
        float m = stats[f] * (1.0f / NN);
        float v = stats[F + f] * (1.0f / NN) - m * m;
        float y = (x[idx] - m) * rsqrtf(v + 1e-5f) * g[f] + be[f];
        x[idx] = fmaxf(y, 0.0f);
    }
}

// ---------------- driver ----------------
static inline int ilog2(int v) { int s = 0; while ((1 << s) < v) ++s; return s; }

extern "C" void kernel_launch(void* const* d_in, const int* in_sizes, int n_in,
                              void* d_out, int out_size, void* d_ws, size_t ws_size,
                              hipStream_t stream)
{
    const float*     x   = (const float*)d_in[0];
    const long long* ei  = (const long long*)d_in[1];
    const float *W1 = (const float*)d_in[2],  *b1  = (const float*)d_in[3];
    const float *W2 = (const float*)d_in[4],  *b2  = (const float*)d_in[5];
    const float *W3 = (const float*)d_in[6],  *b3  = (const float*)d_in[7];
    const float *g1 = (const float*)d_in[8],  *be1 = (const float*)d_in[9];
    const float *g2 = (const float*)d_in[10], *be2 = (const float*)d_in[11];
    const float *g3 = (const float*)d_in[12], *be3 = (const float*)d_in[13];
    const float *Wf1 = (const float*)d_in[14], *bf1 = (const float*)d_in[15];
    const float *Wf2 = (const float*)d_in[16], *bf2 = (const float*)d_in[17];
    const float *Wf3 = (const float*)d_in[18], *bf3 = (const float*)d_in[19];
    const float *Wf4 = (const float*)d_in[20], *bf4 = (const float*)d_in[21];

    float*     bufA  = (float*)d_ws;                  // N x 256 f32
    float*     bufB  = bufA + (size_t)NN * 256;       // N x 256 f32
    float*     dinv  = bufB + (size_t)NN * 256;       // N f32
    float*     stats = dinv + NN;                     // 512 f32
    _Float16*  wt    = (_Float16*)(stats + 512);      // up to 256*256 f16 (128 KB)

    // degree -> dinv
    hipMemsetAsync(dinv, 0, NN * sizeof(float), stream);
    deg_kernel<<<(EE + 255) / 256, 256, 0, stream>>>(ei, dinv);
    dinv_kernel<<<(NN + 255) / 256, 256, 0, stream>>>(dinv);

    #define GEMM(Kc, NOUTc, NPADc, BIASc, RELUc, Ap, Wp, Bp, Cp)                            \
        do {                                                                                \
            int wt_total = (NPADc) * (Kc);                                                  \
            wt_kernel<<<(wt_total + 255) / 256, 256, 0, stream>>>(                          \
                (Wp), wt, (Kc), ilog2(Kc), (NOUTc), wt_total);                              \
            int total = TILES_M * ((NPADc) / 16);                                           \
            wmma_gemm<Kc, NOUTc, NPADc, BIASc, RELUc>                                       \
                <<<(total + 3) / 4, 128, 0, stream>>>((Ap), wt, (Bp), (Cp));                \
        } while (0)

    auto gcn_post = [&](const float* hbuf, float* aggbuf, const float* b,
                        const float* g, const float* be, int F, int shift) {
        selfterm_kernel<<<(NN * F + 255) / 256, 256, 0, stream>>>(hbuf, dinv, b, aggbuf, F, shift);
        long long tot = (long long)EE << shift;
        scatter_kernel<<<(int)((tot + 255) / 256), 256, 0, stream>>>(hbuf, dinv, ei, aggbuf, F, shift);
        hipMemsetAsync(stats, 0, 2 * F * sizeof(float), stream);
        bn_stats_kernel<<<(NN + 511) / 512, F, 0, stream>>>(aggbuf, stats, F);
        bn_norm_relu_kernel<<<(NN * F + 255) / 256, 256, 0, stream>>>(aggbuf, stats, g, be, F, shift);
    };

    // GCN layer 1: 128 -> 32
    GEMM(128, 32, 32, false, false, x, W1, nullptr, bufA);
    gcn_post(bufA, bufB, b1, g1, be1, 32, 5);
    // GCN layer 2: 32 -> 64
    GEMM(32, 64, 64, false, false, bufB, W2, nullptr, bufA);
    gcn_post(bufA, bufB, b2, g2, be2, 64, 6);
    // GCN layer 3: 64 -> 128
    GEMM(64, 128, 128, false, false, bufB, W3, nullptr, bufA);
    gcn_post(bufA, bufB, b3, g3, be3, 128, 7);

    // MLP head
    GEMM(128, 256, 256, true, true,  bufB, Wf1, bf1, bufA);
    GEMM(256, 128, 128, true, true,  bufA, Wf2, bf2, bufB);
    GEMM(128, 64,  64,  true, true,  bufB, Wf3, bf3, bufA);
    GEMM(64,  10,  16,  true, false, bufA, Wf4, bf4, (float*)d_out);

    #undef GEMM
}
